// SelfAttention_63591285785334
// MI455X (gfx1250) — matrix operand order repaired and checked by reading, not compiled, not run on hardware
//
#include <hip/hip_runtime.h>
#include <hip/hip_bf16.h>
#include <math.h>
#include <stdint.h>

#define BATCH 8
#define CH 256
#define NPOS 4096  // 64*64
#define EPS 1e-5f

typedef __bf16 bf16;
typedef __attribute__((ext_vector_type(8)))  bf16 v8bf;
typedef __attribute__((ext_vector_type(16))) bf16 v16bf;
typedef __attribute__((ext_vector_type(8)))  float v8f;
typedef __attribute__((ext_vector_type(4)))  uint32_t v4u;
typedef __attribute__((ext_vector_type(8)))  uint32_t v8u;

static __device__ __forceinline__ v16bf cat8(v8bf lo, v8bf hi) {
  return __builtin_shufflevector(lo, hi, 0,1,2,3,4,5,6,7,8,9,10,11,12,13,14,15);
}

// A fragment (16x32 bf16) from row-major [M][K] with row stride `stride`.
// lane L holds row (L&15); 8-elem chunks at k0=(kk*32+(L>>4)*8) and k0+16.
static __device__ __forceinline__ v16bf load_a(const bf16* m, int stride, int row_base, int kk) {
  const int lane = threadIdx.x & 31;
  const bf16* r = m + (size_t)(row_base + (lane & 15)) * stride + kk * 32 + ((lane >> 4) << 3);
  v8bf lo = *(const v8bf*)(r);
  v8bf hi = *(const v8bf*)(r + 16);
  return cat8(lo, hi);
}

// B fragment (32x16 bf16) from row-major [K][N] with row stride `stride`.
// lane L holds column col0+(L&15); rows k_base+(L>>4)*8+s and k_base+16+(L>>4)*8+s, s=0..7.
static __device__ __forceinline__ v16bf load_b(const bf16* m, int stride, int k_base, int col0) {
  const int lane = threadIdx.x & 31;
  const bf16* r = m + (size_t)(k_base + ((lane >> 4) << 3)) * stride + col0 + (lane & 15);
  v16bf v;
#pragma unroll
  for (int s = 0; s < 8; ++s) {
    v[s] = r[(size_t)s * stride];
    v[s + 8] = r[(size_t)(16 + s) * stride];
  }
  return v;
}

static __device__ __forceinline__ v8f wmma_bf16(v16bf a, v16bf b, v8f c) {
  return __builtin_amdgcn_wmma_f32_16x16x32_bf16(false, a, false, b, (short)0, c, false, false);
}

// Wave-private LDS RAW fence (DS ops are in-order per wave; this stops the
// compiler reordering and conservatively drains DScnt).
static __device__ __forceinline__ void lds_wave_fence() {
  asm volatile("s_wait_dscnt 0" ::: "memory");
}

// TDM: issue a 2-D tiled tensor load (global -> LDS). One DMA op per wave.
// g0: {count=1 | lds_addr | global_addr | type=2}; g1: dims/strides/padding.
static __device__ __forceinline__ void tdm_load_2d(uint64_t gaddr, uint32_t lds_off,
                                                   uint32_t g1w0, uint32_t g1w1,
                                                   uint32_t g1w2, uint32_t g1w3,
                                                   uint32_t g1w4, uint32_t g1w5) {
  v4u g0;
  g0[0] = 1u;                                             // count=1, user mode
  g0[1] = lds_off;                                        // lds_addr
  g0[2] = (uint32_t)gaddr;                                // global_addr[31:0]
  g0[3] = (uint32_t)((gaddr >> 32) & 0x1FFFFFFu) | (2u << 30);  // [56:32] | type=2
  v8u g1 = {g1w0, g1w1, g1w2, g1w3, g1w4, g1w5, 0u, 0u};
  asm volatile("tensor_load_to_lds %0, %1" :: "s"(g0), "s"(g1) : "memory");
}

// Issue the K-tile and V-tile DMAs for key block starting at m0.
//   K tile: 256(c) x 64(m), row 128B padded to 144B  (intv=4: 32DW, amt=3: 4DW)
//   V tile:  64(m) x 256(c), row 512B padded to 528B (intv=6: 128DW, amt=3: 4DW)
static __device__ __forceinline__ void issue_kv_tdm(const bf16* kb, const bf16* vb,
                                                    int m0, uint32_t kt_off, uint32_t vt_off) {
  tdm_load_2d((uint64_t)(uintptr_t)(kb + m0), kt_off,
              (1u << 16) | (1u << 20) | (4u << 22) | (3u << 25),
              4096u << 16,    // tensor_dim0 = 4096 (m, contiguous)
              256u << 16,     // tensor_dim1 = 256  (c)
              64u << 16,      // tile_dim0 = 64
              256u,           // tile_dim1 = 256
              4096u);         // tensor_dim0_stride
  tdm_load_2d((uint64_t)(uintptr_t)(vb + (size_t)m0 * CH), vt_off,
              (1u << 16) | (1u << 20) | (6u << 22) | (3u << 25),
              256u << 16,     // tensor_dim0 = 256  (c, contiguous)
              4096u << 16,    // tensor_dim1 = 4096 (m)
              256u << 16,     // tile_dim0 = 256
              64u,            // tile_dim1 = 64
              256u);          // tensor_dim0_stride
}

// ---------------- Kernel 0: weights fp32 [o][c] -> bf16 transposed [c][o] ----
__global__ __launch_bounds__(256) void prep_weights(
    const float* __restrict__ wq, const float* __restrict__ wk,
    const float* __restrict__ wv, const float* __restrict__ wo,
    bf16* __restrict__ wT) {
  int id = blockIdx.x * 256 + threadIdx.x;
  if (id >= 4 * CH * CH) return;
  int m = id >> 16;
  int i = id & 0xFFFF;
  int o = i >> 8, c = i & 255;
  const float* w = (m == 0) ? wq : (m == 1) ? wk : (m == 2) ? wv : wo;
  wT[(size_t)m * CH * CH + c * CH + o] = (bf16)w[o * CH + c];
}

// ---------------- Kernel 1: GroupNorm(1) stats per batch ---------------------
__global__ __launch_bounds__(256) void gn_stats(const float* __restrict__ x,
                                                float* __restrict__ mean,
                                                float* __restrict__ rstd) {
  const int b = blockIdx.x;
  const float* xb = x + (size_t)b * CH * NPOS;
  float s = 0.f, s2 = 0.f;
  for (int i = threadIdx.x; i < CH * NPOS; i += 256) {
    float v = xb[i];
    s += v; s2 += v * v;
  }
  __shared__ float sh1[256], sh2[256];
  sh1[threadIdx.x] = s; sh2[threadIdx.x] = s2;
  __syncthreads();
  for (int st = 128; st > 0; st >>= 1) {
    if (threadIdx.x < st) {
      sh1[threadIdx.x] += sh1[threadIdx.x + st];
      sh2[threadIdx.x] += sh2[threadIdx.x + st];
    }
    __syncthreads();
  }
  if (threadIdx.x == 0) {
    const float inv_m = 1.0f / (float)(CH * NPOS);
    float mu = sh1[0] * inv_m;
    float var = sh2[0] * inv_m - mu * mu;
    mean[b] = mu;
    rstd[b] = rsqrtf(var + EPS);
  }
}

// ---------------- Kernel 2: fused norm + Q/K/V projection --------------------
// qsT[b][n][o] = (Wq xn + bq) * (log2e / 16)   (bf16, position-major)
// kcm[b][o][n] =  Wk xn + bk                   (bf16, channel-major)
// vT [b][n][o] =  Wv xn + bv                   (bf16, position-major)
__global__ __launch_bounds__(256) void qkv_proj(
    const float* __restrict__ x, const float* __restrict__ gamma,
    const float* __restrict__ beta, const bf16* __restrict__ wT,
    const float* __restrict__ bq, const float* __restrict__ bk,
    const float* __restrict__ bv, const float* __restrict__ mean,
    const float* __restrict__ rstd, bf16* __restrict__ qsT,
    bf16* __restrict__ kcm, bf16* __restrict__ vT) {
  const int b = blockIdx.y;
  const int n0 = blockIdx.x * 64;
  __shared__ bf16 xnT[64 * 264];  // [n][c], padded stride

  const float mu = mean[b], rs = rstd[b];
  {
    const int n = threadIdx.x & 63;
    const int c0 = (threadIdx.x >> 6) * 64;
    const float* xb = x + (size_t)b * CH * NPOS + n0 + n;
    for (int i = 0; i < 64; ++i) {
      int c = c0 + i;
      float v = xb[(size_t)c * NPOS];
      v = (v - mu) * rs * gamma[c] + beta[c];
      xnT[n * 264 + c] = (bf16)v;
    }
  }
  __syncthreads();

  const int w = threadIdx.x >> 5;
  const int lane = threadIdx.x & 31;
  const int col = lane & 15;
  const int rb = (lane >> 4) * 8;
  const int nsub = (w & 3) * 16;
  const int ohalf = (w >> 2) * 128;

  v16bf afr[8];
#pragma unroll
  for (int kk = 0; kk < 8; ++kk) afr[kk] = load_a(xnT, 264, nsub, kk);

  const float qscale = 0.0625f * 1.4426950408889634f;  // (1/sqrt(C)) * log2(e)

#pragma unroll
  for (int m = 0; m < 3; ++m) {  // unrolled -> branchless specialized stores
    const bf16* wTm = wT + (size_t)m * CH * CH;
    const float* bias = (m == 0) ? bq : (m == 1) ? bk : bv;
    for (int ot = 0; ot < 8; ++ot) {
      const int o0 = ohalf + ot * 16;
      v8f acc = {};
#pragma unroll
      for (int kk = 0; kk < 8; ++kk) {
        v16bf bfr = load_b(wTm, CH, kk * 32, o0);
        acc = wmma_bf16(afr[kk], bfr, acc);
      }
      const float bb = bias[o0 + col];
      if (m == 0) {
#pragma unroll
        for (int r = 0; r < 8; ++r) {
          int n = n0 + nsub + rb + r;
          qsT[((size_t)b * NPOS + n) * CH + o0 + col] = (bf16)((acc[r] + bb) * qscale);
        }
      } else if (m == 1) {
        const size_t base = ((size_t)b * CH + o0 + col) * NPOS + n0 + nsub + rb;
#pragma unroll
        for (int r = 0; r < 8; ++r) kcm[base + r] = (bf16)(acc[r] + bb);
      } else {
#pragma unroll
        for (int r = 0; r < 8; ++r) {
          int n = n0 + nsub + rb + r;
          vT[((size_t)b * NPOS + n) * CH + o0 + col] = (bf16)(acc[r] + bb);
        }
      }
    }
  }
}

// ---------------- Kernel 3: flash attention, double-buffered TDM staging -----
// Block = 8 waves, 128 query rows; each wave owns 16 rows end-to-end.
// Two K/V LDS buffers; wave 0 issues the tile jt+1 DMA pair right after the
// barrier that (a) publishes tile jt and (b) proves tile jt-1 reads (same
// buffer as jt+1) are finished. Tensor engine runs concurrently with WMMA.
#define LDS_KT0 0u
#define LDS_VT0 36864u
#define LDS_KT1 70656u
#define LDS_VT1 107520u
#define LDS_P   141312u
// total: 141312 + 8*16*72*2 = 159744 B (<= 320KB per workgroup)

extern __shared__ char smem[];

__global__ __launch_bounds__(256) void attn(
    const bf16* __restrict__ qsT, const bf16* __restrict__ kcm,
    const bf16* __restrict__ vT, bf16* __restrict__ attT) {
  const int b = blockIdx.y;
  const int n0 = blockIdx.x * 128;
  const int w = threadIdx.x >> 5;
  const int lane = threadIdx.x & 31;
  const int col = lane & 15;
  const int hi = lane >> 4;

  bf16* Pw = (bf16*)(smem + LDS_P) + w * (16 * 72);

  const bf16* qb = qsT + (size_t)b * NPOS * CH;
  const bf16* kb = kcm + (size_t)b * CH * NPOS;
  const bf16* vb = vT + (size_t)b * NPOS * CH;

  // this wave's Q fragments (16 rows x 256 channels), resident in VGPRs
  v16bf qa[8];
#pragma unroll
  for (int kk = 0; kk < 8; ++kk) qa[kk] = load_a(qb, CH, n0 + w * 16, kk);

  v8f Oacc[16];
#pragma unroll
  for (int ct = 0; ct < 16; ++ct) Oacc[ct] = (v8f){};
  float rm[8], lsum[8];
#pragma unroll
  for (int r = 0; r < 8; ++r) { rm[r] = -__builtin_inff(); lsum[r] = 0.f; }

  if (threadIdx.x < 32) issue_kv_tdm(kb, vb, 0, LDS_KT0, LDS_VT0);

  for (int jt = 0; jt < NPOS / 64; ++jt) {
    const int m0 = jt * 64;
    if (jt + 2 < NPOS / 64) {  // L2 warm-up for the tile whose DMA issues next
      __builtin_prefetch(kb + (size_t)lane * NPOS + m0 + 128, 0, 0);
      __builtin_prefetch(vb + (size_t)(m0 + 128 + lane) * CH, 0, 0);
    }

    if (threadIdx.x < 32) __builtin_amdgcn_s_wait_tensorcnt(0);  // tile jt landed
    __syncthreads();  // publish tile jt; tile jt-1 reads complete
    if (threadIdx.x < 32 && jt + 1 < NPOS / 64) {
      const int sel = (jt + 1) & 1;
      issue_kv_tdm(kb, vb, m0 + 64, sel ? LDS_KT1 : LDS_KT0, sel ? LDS_VT1 : LDS_VT0);
    }
    const int cb = jt & 1;
    const bf16* Kt = (const bf16*)(smem + (cb ? LDS_KT1 : LDS_KT0));  // [256][72]
    const bf16* Vt = (const bf16*)(smem + (cb ? LDS_VT1 : LDS_VT0));  // [64][264]

    // S = Q^T K tile (16 x 64), already scaled by log2e/sqrt(C) via Q
    v8f z = {};
    v8f s[4] = {z, z, z, z};
#pragma unroll
    for (int kk = 0; kk < 8; ++kk) {
#pragma unroll
      for (int mt = 0; mt < 4; ++mt) {
        v16bf bfr = load_b(Kt, 72, kk * 32, mt * 16);
        s[mt] = wmma_bf16(qa[kk], bfr, s[mt]);
      }
    }

    // online softmax: rows live in (VGPR idx, lane-half); cols across 16 lanes
    float a_scale[8];
#pragma unroll
    for (int r = 0; r < 8; ++r) {
      float tm = fmaxf(fmaxf(s[0][r], s[1][r]), fmaxf(s[2][r], s[3][r]));
      tm = fmaxf(tm, __shfl_xor(tm, 1, 32));
      tm = fmaxf(tm, __shfl_xor(tm, 2, 32));
      tm = fmaxf(tm, __shfl_xor(tm, 4, 32));
      tm = fmaxf(tm, __shfl_xor(tm, 8, 32));
      float mnew = fmaxf(rm[r], tm);
      a_scale[r] = exp2f(rm[r] - mnew);
      rm[r] = mnew;
      float ps = 0.f;
#pragma unroll
      for (int mt = 0; mt < 4; ++mt) {
        float p = exp2f(s[mt][r] - mnew);
        s[mt][r] = p;
        ps += p;
      }
      ps += __shfl_xor(ps, 1, 32);
      ps += __shfl_xor(ps, 2, 32);
      ps += __shfl_xor(ps, 4, 32);
      ps += __shfl_xor(ps, 8, 32);
      lsum[r] = lsum[r] * a_scale[r] + ps;
    }
#pragma unroll
    for (int ct = 0; ct < 16; ++ct)
#pragma unroll
      for (int r = 0; r < 8; ++r) Oacc[ct][r] *= a_scale[r];

    // P (C-layout) -> LDS row-major [16][64] -> A fragments (wave-private)
#pragma unroll
    for (int mt = 0; mt < 4; ++mt)
#pragma unroll
      for (int r = 0; r < 8; ++r)
        Pw[(r + hi * 8) * 72 + mt * 16 + col] = (bf16)s[mt][r];
    lds_wave_fence();

    // O += P * V^T   (k = 64 -> 2 WMMA k-steps, 16 channel tiles)
#pragma unroll
    for (int k2 = 0; k2 < 2; ++k2) {
      v16bf pa = load_a(Pw, 72, 0, k2);
#pragma unroll
      for (int ct = 0; ct < 16; ++ct) {
        v16bf bfr = load_b(Vt, 264, k2 * 32, ct * 16);
        Oacc[ct] = wmma_bf16(pa, bfr, Oacc[ct]);
      }
    }
    lds_wave_fence();  // drain P reads before next-iter overwrite
  }

  float inv[8];
#pragma unroll
  for (int r = 0; r < 8; ++r) inv[r] = 1.0f / lsum[r];
#pragma unroll
  for (int ct = 0; ct < 16; ++ct)
#pragma unroll
    for (int r = 0; r < 8; ++r) {
      int n = n0 + w * 16 + r + hi * 8;
      attT[((size_t)b * NPOS + n) * CH + ct * 16 + col] = (bf16)(Oacc[ct][r] * inv[r]);
    }
}

// ---------------- Kernel 4: output projection + bias + residual --------------
__global__ __launch_bounds__(256) void out_proj(
    const bf16* __restrict__ attT, const bf16* __restrict__ woT,
    const float* __restrict__ bo, const float* __restrict__ x,
    float* __restrict__ out) {
  const int b = blockIdx.y;
  const int n0 = blockIdx.x * 64;
  const int w = threadIdx.x >> 5;
  const int lane = threadIdx.x & 31;
  const int col = lane & 15;
  const int hi = lane >> 4;
  const int nsub = (w & 3) * 16;
  const int ohalf = (w >> 2) * 128;

  const bf16* ab = attT + (size_t)b * NPOS * CH;
  v16bf afr[8];
#pragma unroll
  for (int kk = 0; kk < 8; ++kk) afr[kk] = load_a(ab, CH, n0 + nsub, kk);

  for (int ot = 0; ot < 8; ++ot) {
    const int o0 = ohalf + ot * 16;
    v8f acc = {};
#pragma unroll
    for (int kk = 0; kk < 8; ++kk) {
      v16bf bfr = load_b(woT, CH, kk * 32, o0);
      acc = wmma_bf16(afr[kk], bfr, acc);
    }
    const float bb = bo[o0 + col];
    // per lane: consecutive r = consecutive n at fixed o -> coalesced f32 I/O
    const size_t base = ((size_t)b * CH + o0 + col) * NPOS + n0 + nsub + hi * 8;
#pragma unroll
    for (int r = 0; r < 8; ++r) out[base + r] = x[base + r] + acc[r] + bb;
  }
}

// ---------------- launch -----------------------------------------------------
extern "C" void kernel_launch(void* const* d_in, const int* in_sizes, int n_in,
                              void* d_out, int out_size, void* d_ws, size_t ws_size,
                              hipStream_t stream) {
  const float* x     = (const float*)d_in[0];
  const float* gamma = (const float*)d_in[1];
  const float* beta  = (const float*)d_in[2];
  const float* wq    = (const float*)d_in[3];
  const float* bq    = (const float*)d_in[4];
  const float* wk    = (const float*)d_in[5];
  const float* bk    = (const float*)d_in[6];
  const float* wv    = (const float*)d_in[7];
  const float* bv    = (const float*)d_in[8];
  const float* wo    = (const float*)d_in[9];
  const float* bo    = (const float*)d_in[10];
  float* out = (float*)d_out;

  char* ws = (char*)d_ws;
  bf16* wT = (bf16*)ws;                          // 4 * 256*256 bf16 = 512 KB
  float* mean = (float*)(ws + 524288);           // 8 f32
  float* rstd = mean + 8;                        // 8 f32
  bf16* qsT = (bf16*)(ws + 524288 + 256);        // [B][N][C] bf16, 16 MB
  bf16* kcm = qsT + (size_t)BATCH * CH * NPOS;   // [B][C][N] bf16, 16 MB
  bf16* vT  = kcm + (size_t)BATCH * CH * NPOS;   // [B][N][C] bf16, 16 MB
  bf16* aT  = vT  + (size_t)BATCH * CH * NPOS;   // [B][N][C] bf16, 16 MB

  hipLaunchKernelGGL(prep_weights, dim3(1024), dim3(256), 0, stream, wq, wk, wv, wo, wT);
  hipLaunchKernelGGL(gn_stats, dim3(BATCH), dim3(256), 0, stream, x, mean, rstd);
  hipLaunchKernelGGL(qkv_proj, dim3(NPOS / 64, BATCH), dim3(256), 0, stream,
                     x, gamma, beta, wT, bq, bk, bv, mean, rstd, qsT, kcm, vT);
  hipLaunchKernelGGL(attn, dim3(NPOS / 128, BATCH), dim3(256), 159744, stream,
                     qsT, kcm, vT, aT);
  hipLaunchKernelGGL(out_proj, dim3(NPOS / 64, BATCH), dim3(256), 0, stream,
                     aT, wT + (size_t)3 * CH * CH, bo, x, out);
}
